// SparseAttention3d_2972117369403
// MI455X (gfx1250) — compile-verified, matrix operand order checked by reading, not run on hardware
//
#include <hip/hip_runtime.h>

#define NV 100000
#define MQ 25000
#define KK 48
#define CC 64
#define HH 4
#define DD 16
#define FF 256
#define CO_ 64
#define QB 8
#define RS 72        // padded row stride (f16 elems) for 64-wide tiles
#define FS 264       // padded row stride for 256-wide hidden

typedef __attribute__((ext_vector_type(16))) _Float16 v16h;
typedef __attribute__((ext_vector_type(8)))  _Float16 v8h;
typedef __attribute__((ext_vector_type(8)))  float    v8f;

// ws (f16) weight layout offsets
#define WS_WV   8192       // in_proj rows 128..191
#define WS_OPW  12288      // out_proj_w 64x64
#define WS_L1W  16384      // lin1_w 256x64
#define WS_L2W  32768      // lin2_w 64x256
#define WS_OW   49152      // out_w 64x64
#define WS_TOT  53248

// CDNA5 WMMA f16 A/B fragment from a row-major row: lane L holds row (L&15);
// its 16 f16 values are K = base..base+7 and base+16..base+23,
// base = (L>=16 ? 8 : 0) + kOff.
__device__ __forceinline__ v16h frag_row(const _Float16* row, int kOff, int lane) {
  int kb = ((lane >> 4) << 3) + kOff;
  v8h lo = *(const v8h*)(row + kb);
  v8h hi = *(const v8h*)(row + kb + 16);
  v16h o;
#pragma unroll
  for (int i = 0; i < 8; ++i) { o[i] = lo[i]; o[i + 8] = hi[i]; }
  return o;
}

// Same fragment but gathering a column of a row-major matrix (for B = X^T).
// base points at element [0][n]; stride is the row stride in elements.
__device__ __forceinline__ v16h frag_col(const _Float16* base, int stride, int kOff, int lane) {
  int kb = ((lane >> 4) << 3) + kOff;
  v16h o;
#pragma unroll
  for (int i = 0; i < 8; ++i) {
    o[i]     = base[(kb + i) * stride];
    o[i + 8] = base[(kb + 16 + i) * stride];
  }
  return o;
}

__device__ __forceinline__ v8f wmma16(v16h a, v16h b, v8f c) {
  return __builtin_amdgcn_wmma_f32_16x16x32_f16(false, a, false, b, (short)0, c, false, false);
}

__global__ __launch_bounds__(256)
void prep_weights_f16(const float* __restrict__ in_proj_w,
                      const float* __restrict__ out_proj_w,
                      const float* __restrict__ lin1_w,
                      const float* __restrict__ lin2_w,
                      const float* __restrict__ out_w,
                      _Float16* __restrict__ wh) {
  int i = blockIdx.x * blockDim.x + threadIdx.x;
  if (i < 12288)            wh[i] = (_Float16)in_proj_w[i];
  else if (i < 16384)       wh[i] = (_Float16)out_proj_w[i - 12288];
  else if (i < 32768)       wh[i] = (_Float16)lin1_w[i - 16384];
  else if (i < 49152)       wh[i] = (_Float16)lin2_w[i - 32768];
  else if (i < WS_TOT)      wh[i] = (_Float16)out_w[i - 49152];
}

__global__ __launch_bounds__(256)
void sparse_attn_main(const float* __restrict__ vfeat,
                      const float* __restrict__ vcoord,
                      const float* __restrict__ qcoord,
                      const int* __restrict__ kidx,
                      const unsigned char* __restrict__ kmask,
                      const float* __restrict__ q_pos_w, const float* __restrict__ q_pos_b,
                      const float* __restrict__ k_pos_w, const float* __restrict__ k_pos_b,
                      const float* __restrict__ in_proj_w, const float* __restrict__ in_proj_b,
                      const float* __restrict__ out_proj_b,
                      const float* __restrict__ lin1_b, const float* __restrict__ lin2_b,
                      const float* __restrict__ norm_g, const float* __restrict__ norm_b,
                      const float* __restrict__ norm_m, const float* __restrict__ norm_v,
                      const float* __restrict__ out_b,
                      const float* __restrict__ bn2_g, const float* __restrict__ bn2_b,
                      const float* __restrict__ bn2_m, const float* __restrict__ bn2_v,
                      const _Float16* __restrict__ wh,
                      float* __restrict__ out) {
  extern __shared__ char smem[];
  _Float16* kin  = (_Float16*)smem;            // QB*KK*RS  = 27648 f16
  _Float16* qkp  = kin  + QB * KK * RS;        // QB*16*CC  =  8192 f16 (B for scores, rows>=4 zero)
  _Float16* zmat = qkp  + QB * 16 * CC;        // HH*16*RS  =  4608 f16
  _Float16* ohh  = zmat + HH * 16 * RS;        // 16*RS     =  1152 f16
  _Float16* ah   = ohh  + 16 * RS;             // 16*RS
  _Float16* xh   = ah   + 16 * RS;             // 16*RS
  _Float16* hmat = xh   + 16 * RS;             // 16*FS     =  4224 f16
  _Float16* ap   = hmat + 16 * FS;             // QB*16*CC  =  8192 f16 (probs, K-padded w/ zeros)
  float* sprob = (float*)(ap + QB * 16 * CC);  // QB*HH*KK  =  1536 f32
  float* qin   = sprob + QB * HH * KK;         // QB*CC
  float* qfs   = qin   + QB * CC;              // QB*CC (scaled q)
  float* attf  = qfs   + QB * CC;              // QB*CC (attend, fp32 residual)
  float* rel   = attf  + QB * CC;              // QB*KK*3
  int*   idxl  = (int*)(rel + QB * KK * 3);    // QB*KK
  float* sb    = (float*)(idxl + QB * KK);     // QB*HH  (q·bk score bias)

  const int tid  = threadIdx.x;
  const int lane = tid & 31;
  const int wid  = tid >> 5;
  const int q0   = blockIdx.x * QB;

  const _Float16* wvh = wh + WS_WV;
  const _Float16* wop = wh + WS_OPW;
  const _Float16* wl1 = wh + WS_L1W;
  const _Float16* wl2 = wh + WS_L2W;
  const _Float16* wow = wh + WS_OW;

  // ---- P0: gather indices, relative coords, q_in; zero padded B/z tiles ----
  for (int p = tid; p < QB * KK; p += 256) {
    int qi = p / KK, j = p - qi * KK;
    int gi = kidx[(q0 + qi) * KK + j];
    idxl[p] = gi;
    __builtin_prefetch(vfeat + (long)gi * CC, 0, 1);   // warm L2/L0 for P1 gathers
    rel[p * 3 + 0] = vcoord[gi * 3 + 0] - qcoord[(q0 + qi) * 3 + 0];
    rel[p * 3 + 1] = vcoord[gi * 3 + 1] - qcoord[(q0 + qi) * 3 + 1];
    rel[p * 3 + 2] = vcoord[gi * 3 + 2] - qcoord[(q0 + qi) * 3 + 2];
  }
  for (int e = tid; e < QB * CC; e += 256) {
    int qi = e >> 6, c = e & 63;
    const float* qc = qcoord + (q0 + qi) * 3;
    float acc = q_pos_b[c] + qc[0] * q_pos_w[c * 3] + qc[1] * q_pos_w[c * 3 + 1]
              + qc[2] * q_pos_w[c * 3 + 2];
    qin[e] = fmaxf(acc, 0.f);
  }
  for (int e = tid; e < QB * 16 * CC; e += 256) qkp[e] = (_Float16)0.f;
  for (int e = tid; e < HH * 16 * RS; e += 256) zmat[e] = (_Float16)0.f;
  __syncthreads();

  // ---- P1: k_in = gathered feats + relu(pos MLP)  (f16, LDS); scaled q ----
  for (int e = tid; e < QB * KK * CC; e += 256) {
    int p = e >> 6, c = e & 63;
    float kp = k_pos_b[c] + rel[p * 3] * k_pos_w[c * 3]
             + rel[p * 3 + 1] * k_pos_w[c * 3 + 1]
             + rel[p * 3 + 2] * k_pos_w[c * 3 + 2];
    kp = fmaxf(kp, 0.f);
    float kf = vfeat[(long)idxl[p] * CC + c];
    kin[p * RS + c] = (_Float16)(kf + kp);
  }
  for (int e = tid; e < QB * CC; e += 256) {      // q = (q_in@Wq^T + bq)*d^-0.5
    int qi = e >> 6, hc = e & 63;
    float acc = in_proj_b[hc];
    const float* wr = in_proj_w + hc * CC;
    const float* x  = qin + qi * CC;
#pragma unroll 8
    for (int c = 0; c < CC; ++c) acc += x[c] * wr[c];
    qfs[e] = acc * 0.25f;                          // d=16 -> 1/sqrt(d)=0.25
  }
  __syncthreads();

  // ---- P2: fold Wk into query: qk[m,h,:] = Wk_h^T q_h ; score bias q·bk ----
  for (int e = tid; e < QB * HH * CC; e += 256) {
    int qi = e / (HH * CC); int r = e - qi * HH * CC; int h = r >> 6; int c = r & 63;
    float acc = 0.f;
#pragma unroll
    for (int d = 0; d < DD; ++d)
      acc += in_proj_w[(CC + h * DD + d) * CC + c] * qfs[qi * CC + h * DD + d];
    qkp[(qi * 16 + h) * CC + c] = (_Float16)acc;
  }
  if (tid < QB * HH) {
    int qi = tid >> 2, h = tid & 3;
    float acc = 0.f;
#pragma unroll
    for (int d = 0; d < DD; ++d)
      acc += qfs[qi * CC + h * DD + d] * in_proj_b[CC + h * DD + d];
    sb[tid] = acc;
  }
  __syncthreads();

  // ---- P3: scores via WMMA: (48x64 k_in) x (64x16 qk-pad) per query ----
  for (int job = wid; job < QB * 3; job += 8) {
    int qi = job / 3, mt = job - qi * 3;
    v8f acc = {};
    const _Float16* arow = kin + ((qi * KK + mt * 16 + (lane & 15))) * RS;
    const _Float16* brow = qkp + (qi * 16 + (lane & 15)) * CC;
#pragma unroll
    for (int kc = 0; kc < 2; ++kc)
      acc = wmma16(frag_row(arow, kc * 32, lane), frag_row(brow, kc * 32, lane), acc);
    int n = lane & 15, mo = (lane >> 4) << 3;
    if (n < HH) {
      float sbv = sb[qi * HH + n];
#pragma unroll
      for (int i = 0; i < 8; ++i)
        sprob[qi * HH * KK + n * KK + mt * 16 + mo + i] = acc[i] + sbv;
    }
  }
  __syncthreads();

  // ---- P4: masked softmax over K=48; emit f16 prob A-tiles (K-pad zeroed) ----
  if (tid < QB * HH) {
    int qi = tid >> 2, h = tid & 3;
    float* sr = sprob + qi * HH * KK + h * KK;
    const unsigned char* mr = kmask + (q0 + qi) * KK;
    float mx = -1e30f;
    for (int j = 0; j < KK; ++j) { float v = mr[j] ? -1e9f : sr[j]; sr[j] = v; mx = fmaxf(mx, v); }
    float sum = 0.f;
    for (int j = 0; j < KK; ++j) { float e = __expf(sr[j] - mx); sr[j] = e; sum += e; }
    float inv = 1.f / sum;
    _Float16* arow = ap + (qi * 16 + h) * CC;
    for (int j = 0; j < KK; ++j) arow[j] = (_Float16)(sr[j] * inv);
    for (int j = KK; j < CC; ++j) arow[j] = (_Float16)0.f;   // zero K-pad
  }
  __syncthreads();

  // ---- P5: z = a @ k_in via WMMA (A = padded probs, B = k_in^T col-gather) ----
  for (int job = wid; job < QB * 4; job += 8) {
    int qi = job >> 2, nt = job & 3;
    v8f acc = {};
    const _Float16* arow = ap + (qi * 16 + (lane & 15)) * CC;
    const _Float16* bcol = kin + qi * KK * RS + nt * 16 + (lane & 15);
#pragma unroll
    for (int kc = 0; kc < 2; ++kc)
      acc = wmma16(frag_row(arow, kc * 32, lane), frag_col(bcol, RS, kc * 32, lane), acc);
    int n = lane & 15, mo = (lane >> 4) << 3;
    if (mo == 0) {                                  // rows M = h, only h < 4 valid
#pragma unroll
      for (int i = 0; i < HH; ++i)
        zmat[(i * 16 + qi) * RS + nt * 16 + n] = (_Float16)acc[i];
    }
  }
  __syncthreads();

  // ---- P6: o = z @ Wv_h^T + bv  (block-diagonal: one 16x16 tile per head) ----
  for (int job = wid; job < HH; job += 8) {
    int h = job;
    v8f acc = {};
    const _Float16* arow = zmat + (h * 16 + (lane & 15)) * RS;
    const _Float16* brow = wvh + (h * DD + (lane & 15)) * CC;
#pragma unroll
    for (int kc = 0; kc < 2; ++kc)
      acc = wmma16(frag_row(arow, kc * 32, lane), frag_row(brow, kc * 32, lane), acc);
    int n = lane & 15, mo = (lane >> 4) << 3;
    float bv = in_proj_b[2 * CC + h * DD + n];      // softmax sums to 1 -> bias adds directly
#pragma unroll
    for (int i = 0; i < 8; ++i)
      ohh[(mo + i) * RS + h * DD + n] = (_Float16)(acc[i] + bv);
  }
  __syncthreads();

  // ---- P7: attend = o @ out_proj^T + b ----
  for (int job = wid; job < 4; job += 8) {
    int nt = job;
    v8f acc = {};
    const _Float16* arow = ohh + (lane & 15) * RS;
    const _Float16* brow = wop + (nt * 16 + (lane & 15)) * CC;
#pragma unroll
    for (int kc = 0; kc < 2; ++kc)
      acc = wmma16(frag_row(arow, kc * 32, lane), frag_row(brow, kc * 32, lane), acc);
    int n = lane & 15, mo = (lane >> 4) << 3;
    int c = nt * 16 + n;
    float bias = out_proj_b[c];
#pragma unroll
    for (int i = 0; i < 8; ++i) {
      int qi = mo + i;
      float v = acc[i] + bias;
      ah[qi * RS + c] = (_Float16)v;
      if (qi < QB) attf[qi * CC + c] = v;
    }
  }
  __syncthreads();

  // ---- P8: hidden = relu(attend @ lin1^T + b)  (16 N-tiles) ----
  for (int job = wid; job < FF / 16; job += 8) {
    int nt = job;
    v8f acc = {};
    const _Float16* arow = ah + (lane & 15) * RS;
    const _Float16* brow = wl1 + (nt * 16 + (lane & 15)) * CC;
#pragma unroll
    for (int kc = 0; kc < 2; ++kc)
      acc = wmma16(frag_row(arow, kc * 32, lane), frag_row(brow, kc * 32, lane), acc);
    int n = lane & 15, mo = (lane >> 4) << 3;
    int f = nt * 16 + n;
    float bias = lin1_b[f];
#pragma unroll
    for (int i = 0; i < 8; ++i)
      hmat[(mo + i) * FS + f] = (_Float16)fmaxf(acc[i] + bias, 0.f);
  }
  __syncthreads();

  // ---- P9: x = BN(attend + hidden @ lin2^T + b)  (K=256 -> 8 WMMA steps) ----
  for (int job = wid; job < 4; job += 8) {
    int nt = job;
    v8f acc = {};
    const _Float16* arow = hmat + (lane & 15) * FS;
    const _Float16* brow = wl2 + (nt * 16 + (lane & 15)) * FF;
#pragma unroll
    for (int kc = 0; kc < 8; ++kc)
      acc = wmma16(frag_row(arow, kc * 32, lane), frag_row(brow, kc * 32, lane), acc);
    int n = lane & 15, mo = (lane >> 4) << 3;
    int c = nt * 16 + n;
    float bias = lin2_b[c];
    float g = norm_g[c], bb = norm_b[c], mm = norm_m[c];
    float inv = rsqrtf(norm_v[c] + 1e-5f);
#pragma unroll
    for (int i = 0; i < 8; ++i) {
      int qi = mo + i;
      float att = (qi < QB) ? attf[qi * CC + c] : 0.f;
      float x = att + acc[i] + bias;
      x = (x - mm) * inv * g + bb;
      xh[qi * RS + c] = (_Float16)x;
    }
  }
  __syncthreads();

  // ---- P10: y = relu(BN2(x @ out_w^T + b)) -> global ----
  for (int job = wid; job < 4; job += 8) {
    int nt = job;
    v8f acc = {};
    const _Float16* arow = xh + (lane & 15) * RS;
    const _Float16* brow = wow + (nt * 16 + (lane & 15)) * CC;
#pragma unroll
    for (int kc = 0; kc < 2; ++kc)
      acc = wmma16(frag_row(arow, kc * 32, lane), frag_row(brow, kc * 32, lane), acc);
    int n = lane & 15, mo = (lane >> 4) << 3;
    int co = nt * 16 + n;
    float bias = out_b[co];
    float g = bn2_g[co], bb = bn2_b[co], mm = bn2_m[co];
    float inv = rsqrtf(bn2_v[co] + 1e-5f);
#pragma unroll
    for (int i = 0; i < 8; ++i) {
      int qi = mo + i;
      if (qi < QB) {
        float y = acc[i] + bias;
        y = (y - mm) * inv * g + bb;
        out[(long)(q0 + qi) * CO_ + co] = fmaxf(y, 0.f);
      }
    }
  }
}

extern "C" void kernel_launch(void* const* d_in, const int* in_sizes, int n_in,
                              void* d_out, int out_size, void* d_ws, size_t ws_size,
                              hipStream_t stream) {
  (void)in_sizes; (void)n_in; (void)out_size; (void)ws_size;
  const float* vfeat      = (const float*)d_in[0];
  const float* vcoord     = (const float*)d_in[1];
  const float* qcoord     = (const float*)d_in[2];
  const int*   kidx       = (const int*)d_in[3];
  const unsigned char* km = (const unsigned char*)d_in[4];
  const float* q_pos_w    = (const float*)d_in[5];
  const float* q_pos_b    = (const float*)d_in[6];
  const float* k_pos_w    = (const float*)d_in[7];
  const float* k_pos_b    = (const float*)d_in[8];
  const float* in_proj_w  = (const float*)d_in[9];
  const float* in_proj_b  = (const float*)d_in[10];
  const float* out_proj_w = (const float*)d_in[11];
  const float* out_proj_b = (const float*)d_in[12];
  const float* lin1_w     = (const float*)d_in[13];
  const float* lin1_b     = (const float*)d_in[14];
  const float* lin2_w     = (const float*)d_in[15];
  const float* lin2_b     = (const float*)d_in[16];
  const float* norm_g     = (const float*)d_in[17];
  const float* norm_b     = (const float*)d_in[18];
  const float* norm_m     = (const float*)d_in[19];
  const float* norm_v     = (const float*)d_in[20];
  const float* out_w      = (const float*)d_in[21];
  const float* out_b      = (const float*)d_in[22];
  const float* bn2_g      = (const float*)d_in[23];
  const float* bn2_b      = (const float*)d_in[24];
  const float* bn2_m      = (const float*)d_in[25];
  const float* bn2_v      = (const float*)d_in[26];

  _Float16* wh = (_Float16*)d_ws;
  prep_weights_f16<<<(WS_TOT + 255) / 256, 256, 0, stream>>>(
      in_proj_w, out_proj_w, lin1_w, lin2_w, out_w, wh);

  // dynamic LDS: f16 region + f32 region + int region + score-bias
  size_t f16elems = (size_t)QB * KK * RS + 2 * (QB * 16 * CC) + HH * 16 * RS
                  + 3 * 16 * RS + 16 * FS;
  size_t f32elems = (size_t)QB * HH * KK + 3 * QB * CC + QB * KK * 3;
  size_t smem = f16elems * 2 + f32elems * 4 + (size_t)QB * KK * 4 + (size_t)QB * HH * 4;

  sparse_attn_main<<<MQ / QB, 256, smem, stream>>>(
      vfeat, vcoord, qcoord, kidx, km,
      q_pos_w, q_pos_b, k_pos_w, k_pos_b,
      in_proj_w, in_proj_b, out_proj_b,
      lin1_b, lin2_b, norm_g, norm_b, norm_m, norm_v,
      out_b, bn2_g, bn2_b, bn2_m, bn2_v,
      wh, (float*)d_out);
}
